// biGCNLayer_67259187855859
// MI455X (gfx1250) — matrix operand order repaired and checked by reading, not compiled, not run on hardware
//
#include <hip/hip_runtime.h>
#include <hip/hip_bf16.h>
#include <cstdint>
#include <cstddef>

typedef __bf16 bf16_t;
typedef bf16_t v16bf __attribute__((ext_vector_type(16)));
typedef float  v8f   __attribute__((ext_vector_type(8)));

union BF16Frag {
  v16bf v;
  uint4 q[2];
  bf16_t e[16];
};

union F16x {
  float4 v[4];
  float f[16];
};

__device__ __forceinline__ v8f wmma_bf16(v16bf a, v16bf b, v8f c) {
  // D = A(16x32 bf16) x B(32x16 bf16) + C(16x16 f32)
  return __builtin_amdgcn_wmma_f32_16x16x32_bf16(false, a, false, b, (short)0, c, false, false);
}

// ---------------------------------------------------------------------------
// prep kernels
// ---------------------------------------------------------------------------
__global__ void zero_kernel(float* __restrict__ p, size_t n) {
  size_t i = (size_t)blockIdx.x * blockDim.x + threadIdx.x;
  size_t stride = (size_t)gridDim.x * blockDim.x;
  for (; i < n; i += stride) p[i] = 0.0f;
}

__global__ void cvt_bf16_kernel(const float* __restrict__ in, bf16_t* __restrict__ out, size_t n) {
  size_t i = (size_t)blockIdx.x * blockDim.x + threadIdx.x;
  size_t stride = (size_t)gridDim.x * blockDim.x;
  for (; i < n; i += stride) out[i] = (bf16_t)in[i];
}

// W is [K x N] row-major f32; Wt is [N x K] row-major bf16 (i.e., transposed)
__global__ void transpose_cvt_kernel(const float* __restrict__ W, bf16_t* __restrict__ Wt,
                                     int K, int N) {
  int idx = blockIdx.x * blockDim.x + threadIdx.x;
  if (idx < K * N) {
    int n = idx / K;
    int k = idx - n * K;
    Wt[idx] = (bf16_t)W[(size_t)k * N + n];
  }
}

// ---------------------------------------------------------------------------
// generic WMMA GEMM: C[M x NCOLS] = act(A[M x K] @ Bt[NCOLS x K]^T + bias)
// block = (NCOLS/32) waves; each wave owns a 16x32 column slab; 16 rows/block
// ---------------------------------------------------------------------------
template<int NCOLS, int K, bool RELU, bool OUTBF>
__global__ void __launch_bounds__(256)
wmma_gemm_kernel(const bf16_t* __restrict__ A, const bf16_t* __restrict__ Bt,
                 const float* __restrict__ bias, int M, void* __restrict__ Cout) {
  const int wave = threadIdx.x >> 5;
  const int lane = threadIdx.x & 31;
  const int half = lane >> 4;
  const int lr   = lane & 15;
  const int row0 = blockIdx.x * 16;
  const int n0 = wave * 32;
  const int n1 = n0 + 16;

  const int mA = min(row0 + lr, M - 1);           // clamp; stores are guarded
  const bf16_t* arow = A + (size_t)mA * K;
  const int sA = half ? 8 : 0;                    // A K-offset pattern per half
  const int kB = half ? 8 : 0;                    // B K-offset pattern per half

  v8f acc0 = {}; v8f acc1 = {};
#pragma unroll
  for (int kc = 0; kc < K / 32; ++kc) {
    BF16Frag af;
    af.q[0] = *reinterpret_cast<const uint4*>(arow + kc * 32 + sA);       // K: +{0..7}
    af.q[1] = *reinterpret_cast<const uint4*>(arow + kc * 32 + sA + 16);  // K: +{16..23}
    BF16Frag b0, b1;
    const bf16_t* bp0 = Bt + (size_t)(n0 + lr) * K + kc * 32 + kB;
    const bf16_t* bp1 = Bt + (size_t)(n1 + lr) * K + kc * 32 + kB;
    b0.q[0] = reinterpret_cast<const uint4*>(bp0)[0];
    b0.q[1] = reinterpret_cast<const uint4*>(bp0)[2];
    b1.q[0] = reinterpret_cast<const uint4*>(bp1)[0];
    b1.q[1] = reinterpret_cast<const uint4*>(bp1)[2];
    acc0 = wmma_bf16(af.v, b0.v, acc0);
    acc1 = wmma_bf16(af.v, b1.v, acc1);
  }

  const float bia0 = bias[n0 + lr];
  const float bia1 = bias[n1 + lr];
#pragma unroll
  for (int r = 0; r < 8; ++r) {
    int m = row0 + r + 8 * half;
    if (m < M) {
      float v0 = acc0[r] + bia0;
      float v1 = acc1[r] + bia1;
      if (RELU) { v0 = fmaxf(v0, 0.0f); v1 = fmaxf(v1, 0.0f); }
      if constexpr (OUTBF) {
        bf16_t* C = (bf16_t*)Cout;
        C[(size_t)m * NCOLS + n0 + lr] = (bf16_t)v0;
        C[(size_t)m * NCOLS + n1 + lr] = (bf16_t)v1;
      } else {
        float* C = (float*)Cout;
        C[(size_t)m * NCOLS + n0 + lr] = v0;
        C[(size_t)m * NCOLS + n1 + lr] = v1;
      }
    }
  }
}

// ---------------------------------------------------------------------------
// fused: logits = H1 @ Wt_s2^T + b_s2 ; Sh = softmax(logits, axis=1)
// block = 4 waves (128 cols), 16 rows; softmax via LDS
// ---------------------------------------------------------------------------
__global__ void __launch_bounds__(128)
s2_softmax_kernel(const bf16_t* __restrict__ A, const bf16_t* __restrict__ Bt,
                  const float* __restrict__ bias, int M, float* __restrict__ ShF) {
  constexpr int K = 256, NC = 128;
  __shared__ float sm[16 * NC];
  __shared__ float red[16 * 8];
  __shared__ float rowv[16];

  const int wave = threadIdx.x >> 5;
  const int lane = threadIdx.x & 31;
  const int half = lane >> 4;
  const int lr   = lane & 15;
  const int row0 = blockIdx.x * 16;
  const int n0 = wave * 32;
  const int n1 = n0 + 16;

  const int mA = min(row0 + lr, M - 1);
  const bf16_t* arow = A + (size_t)mA * K;
  const int sA = half ? 8 : 0;
  const int kB = half ? 8 : 0;

  v8f acc0 = {}; v8f acc1 = {};
#pragma unroll
  for (int kc = 0; kc < K / 32; ++kc) {
    BF16Frag af;
    af.q[0] = *reinterpret_cast<const uint4*>(arow + kc * 32 + sA);
    af.q[1] = *reinterpret_cast<const uint4*>(arow + kc * 32 + sA + 16);
    BF16Frag b0, b1;
    const bf16_t* bp0 = Bt + (size_t)(n0 + lr) * K + kc * 32 + kB;
    const bf16_t* bp1 = Bt + (size_t)(n1 + lr) * K + kc * 32 + kB;
    b0.q[0] = reinterpret_cast<const uint4*>(bp0)[0];
    b0.q[1] = reinterpret_cast<const uint4*>(bp0)[2];
    b1.q[0] = reinterpret_cast<const uint4*>(bp1)[0];
    b1.q[1] = reinterpret_cast<const uint4*>(bp1)[2];
    acc0 = wmma_bf16(af.v, b0.v, acc0);
    acc1 = wmma_bf16(af.v, b1.v, acc1);
  }

  const float bia0 = bias[n0 + lr];
  const float bia1 = bias[n1 + lr];
#pragma unroll
  for (int r = 0; r < 8; ++r) {
    int mrow = r + 8 * half;
    sm[mrow * NC + n0 + lr] = acc0[r] + bia0;
    sm[mrow * NC + n1 + lr] = acc1[r] + bia1;
  }
  __syncthreads();

  // softmax: 8 threads per row, 16 cols each
  const int r  = threadIdx.x >> 3;
  const int cg = threadIdx.x & 7;
  float* rowp = &sm[r * NC + cg * 16];

  float mx = -3.402823e38f;
#pragma unroll
  for (int i = 0; i < 16; ++i) mx = fmaxf(mx, rowp[i]);
  red[r * 8 + cg] = mx;
  __syncthreads();
  if (cg == 0) {
    float m2 = red[r * 8];
#pragma unroll
    for (int j = 1; j < 8; ++j) m2 = fmaxf(m2, red[r * 8 + j]);
    rowv[r] = m2;
  }
  __syncthreads();
  const float rm = rowv[r];
  float s = 0.0f;
#pragma unroll
  for (int i = 0; i < 16; ++i) { float ev = __expf(rowp[i] - rm); rowp[i] = ev; s += ev; }
  red[r * 8 + cg] = s;
  __syncthreads();
  if (cg == 0) {
    float s2 = 0.0f;
#pragma unroll
    for (int j = 0; j < 8; ++j) s2 += red[r * 8 + j];
    rowv[r] = s2;
  }
  __syncthreads();
  const float inv = 1.0f / rowv[r];
  const int grow = row0 + r;
  if (grow < M) {
#pragma unroll
    for (int i = 0; i < 16; ++i) {
      ShF[(size_t)grow * NC + cg * 16 + i] = rowp[i] * inv;
    }
  }
}

// ---------------------------------------------------------------------------
// edge metric: Sd = Sh[src]-Sh[dst]; Sd_h = Sd @ Wm + bm; sig = sigmoid(exp(-|Sd_h|/2))
// one wave = 16 edges; 4 k-chunks x 8 n-tiles = 32 WMMAs/wave
// ---------------------------------------------------------------------------
__global__ void __launch_bounds__(256)
edge_metric_kernel(const float* __restrict__ Sh, const bf16_t* __restrict__ WtM,
                   const float* __restrict__ bM, const int* __restrict__ src,
                   const int* __restrict__ dst, float* __restrict__ sigma, int E) {
  const int wave = threadIdx.x >> 5;
  const int lane = threadIdx.x & 31;
  const int half = lane >> 4;
  const int lr   = lane & 15;
  const int e0 = (blockIdx.x * 8 + wave) * 16;
  const int e  = min(e0 + lr, E - 1);

  const float* srow = Sh + (size_t)src[e] * 128;
  const float* drow = Sh + (size_t)dst[e] * 128;
  const int sA = half ? 8 : 0;
  const int kB = half ? 8 : 0;

  v8f zero = {};
  v8f acc[8];
#pragma unroll
  for (int t = 0; t < 8; ++t) acc[t] = zero;

#pragma unroll
  for (int kc = 0; kc < 4; ++kc) {
    const int base = kc * 32 + sA;
    F16x sa, da;
    sa.v[0] = *reinterpret_cast<const float4*>(srow + base);
    sa.v[1] = *reinterpret_cast<const float4*>(srow + base + 4);
    sa.v[2] = *reinterpret_cast<const float4*>(srow + base + 16);
    sa.v[3] = *reinterpret_cast<const float4*>(srow + base + 20);
    da.v[0] = *reinterpret_cast<const float4*>(drow + base);
    da.v[1] = *reinterpret_cast<const float4*>(drow + base + 4);
    da.v[2] = *reinterpret_cast<const float4*>(drow + base + 16);
    da.v[3] = *reinterpret_cast<const float4*>(drow + base + 20);
    BF16Frag af;
#pragma unroll
    for (int i = 0; i < 16; ++i) af.e[i] = (bf16_t)(sa.f[i] - da.f[i]);

#pragma unroll
    for (int t = 0; t < 8; ++t) {
      BF16Frag bf;
      const bf16_t* bp = WtM + (size_t)(t * 16 + lr) * 128 + kc * 32 + kB;
      bf.q[0] = reinterpret_cast<const uint4*>(bp)[0];
      bf.q[1] = reinterpret_cast<const uint4*>(bp)[2];
      acc[t] = wmma_bf16(af.v, bf.v, acc[t]);
    }
  }

  float ps[8];
#pragma unroll
  for (int r = 0; r < 8; ++r) ps[r] = 0.0f;
#pragma unroll
  for (int t = 0; t < 8; ++t) {
    const float bj = bM[t * 16 + lr];
#pragma unroll
    for (int r = 0; r < 8; ++r) { float v = acc[t][r] + bj; ps[r] += v * v; }
  }
  // reduce over the 16 lanes of each half (halves hold disjoint edge rows)
#pragma unroll
  for (int mask = 1; mask <= 8; mask <<= 1) {
#pragma unroll
    for (int r = 0; r < 8; ++r) ps[r] += __shfl_xor(ps[r], mask, 16);
  }
  if (lr == 0) {
#pragma unroll
    for (int r = 0; r < 8; ++r) {
      int ee = e0 + r + 8 * half;
      if (ee < E) {
        float D  = sqrtf(ps[r]);
        float gd = __expf(-0.5f * D);          // SIGMA = 1.0
        sigma[ee] = 1.0f / (1.0f + __expf(-gd));
      }
    }
  }
}

// ---------------------------------------------------------------------------
// scatter: sum_msg[dst] += feature[src]*sig ; deg[dst] += 1 ; sum_sig[dst] += sig
// ---------------------------------------------------------------------------
__global__ void __launch_bounds__(256)
scatter_kernel(const float* __restrict__ feat, const int* __restrict__ src,
               const int* __restrict__ dst, const float* __restrict__ sig,
               float* __restrict__ sum_msg, float* __restrict__ deg,
               float* __restrict__ sum_sig, int E) {
  constexpr int EPB = 32;
  __shared__ int   s_src[EPB];
  __shared__ int   s_dst[EPB];
  __shared__ float s_sig[EPB];
  const int base = blockIdx.x * EPB;
  const int t = threadIdx.x;
  if (t < EPB) {
    int e = base + t;
    if (e < E) { s_src[t] = src[e]; s_dst[t] = dst[e]; s_sig[t] = sig[e]; }
  }
  __syncthreads();
  for (int i = 0; i < EPB; ++i) {
    if (base + i >= E) break;
    const int u = s_src[i];
    const int v = s_dst[i];
    const float sg = s_sig[i];
    atomicAdd(&sum_msg[(size_t)v * 256 + t], feat[(size_t)u * 256 + t] * sg);
    if (t == 0) {
      atomicAdd(&deg[v], 1.0f);
      atomicAdd(&sum_sig[v], sg);
    }
  }
}

// h0 = (sum_msg / max(deg,1)) / (sum_sig + 1e-6) -> bf16
__global__ void h0_kernel(const float* __restrict__ sum_msg, const float* __restrict__ deg,
                          const float* __restrict__ sum_sig, bf16_t* __restrict__ h0, int M) {
  size_t total = (size_t)M * 256;
  size_t i = (size_t)blockIdx.x * blockDim.x + threadIdx.x;
  size_t stride = (size_t)gridDim.x * blockDim.x;
  for (; i < total; i += stride) {
    int r = (int)(i >> 8);
    float d = fmaxf(deg[r], 1.0f);
    float v = (sum_msg[i] / d) / (sum_sig[r] + 1e-6f);
    h0[i] = (bf16_t)v;
  }
}

// ---------------------------------------------------------------------------
// batch norm
// ---------------------------------------------------------------------------
__global__ void __launch_bounds__(256)
bn_reduce_kernel(const float* __restrict__ h, float* __restrict__ colsum,
                 float* __restrict__ colsq, int M) {
  const int c = threadIdx.x;
  float s = 0.0f, ss = 0.0f;
  for (int r = blockIdx.x; r < M; r += gridDim.x) {
    float v = h[(size_t)r * 256 + c];
    s += v;
    ss += v * v;
  }
  atomicAdd(&colsum[c], s);
  atomicAdd(&colsq[c], ss);
}

__global__ void bn_finalize_kernel(const float* __restrict__ colsum, const float* __restrict__ colsq,
                                   const float* __restrict__ gamma, const float* __restrict__ beta,
                                   int M, float* __restrict__ scale, float* __restrict__ shift) {
  const int c = threadIdx.x;
  const float invM = 1.0f / (float)M;
  float mean = colsum[c] * invM;
  float var  = colsq[c] * invM - mean * mean;
  float inv  = rsqrtf(var + 1e-5f);
  float sc   = inv * gamma[c];
  scale[c] = sc;
  shift[c] = beta[c] - mean * sc;
}

__global__ void bn_apply_kernel(float* __restrict__ h, const float* __restrict__ scale,
                                const float* __restrict__ shift, size_t total) {
  size_t i = (size_t)blockIdx.x * blockDim.x + threadIdx.x;
  size_t stride = (size_t)gridDim.x * blockDim.x;
  for (; i < total; i += stride) {
    int c = (int)(i & 255);
    float v = h[i] * scale[c] + shift[c];
    h[i] = fmaxf(v, 0.0f);
  }
}

// ---------------------------------------------------------------------------
// host launch
// ---------------------------------------------------------------------------
extern "C" void kernel_launch(void* const* d_in, const int* in_sizes, int n_in,
                              void* d_out, int out_size, void* d_ws, size_t ws_size,
                              hipStream_t stream) {
  (void)n_in; (void)out_size; (void)ws_size;
  constexpr int IN = 256, OUT = 256, AD = 128;

  const float* feature  = (const float*)d_in[0];
  const float* W_s1     = (const float*)d_in[1];
  const float* b_s1     = (const float*)d_in[2];
  const float* W_s2     = (const float*)d_in[3];
  const float* b_s2     = (const float*)d_in[4];
  const float* W_metric = (const float*)d_in[5];
  const float* b_metric = (const float*)d_in[6];
  const float* W_apply  = (const float*)d_in[7];
  const float* b_apply  = (const float*)d_in[8];
  const float* bn_gamma = (const float*)d_in[9];
  const float* bn_beta  = (const float*)d_in[10];
  const int*   src      = (const int*)d_in[11];
  const int*   dst      = (const int*)d_in[12];

  const int M = in_sizes[0] / IN;   // 50000 nodes
  const int E = in_sizes[11];       // 400000 edges

  // bump allocator over workspace
  char* ws = (char*)d_ws;
  size_t off = 0;
  auto alloc = [&](size_t bytes) -> char* {
    char* p = ws + off;
    off += (bytes + 255) & ~(size_t)255;
    return p;
  };
  bf16_t* featb = (bf16_t*)alloc((size_t)M * IN * 2);
  bf16_t* h1b   = (bf16_t*)alloc((size_t)M * OUT * 2);   // reused later as h0
  bf16_t* wt1   = (bf16_t*)alloc((size_t)IN * OUT * 2);
  bf16_t* wt2   = (bf16_t*)alloc((size_t)OUT * AD * 2);
  bf16_t* wtm   = (bf16_t*)alloc((size_t)AD * AD * 2);
  bf16_t* wta   = (bf16_t*)alloc((size_t)IN * OUT * 2);
  float* sigma  = (float*)alloc((size_t)E * 4);
  float* summsg = (float*)alloc((size_t)M * OUT * 4);
  float* degv   = (float*)alloc((size_t)M * 4);
  float* sumsig = (float*)alloc((size_t)M * 4);
  float* colsum = (float*)alloc(256 * 4);
  float* colsq  = (float*)alloc(256 * 4);
  float* scale  = (float*)alloc(256 * 4);
  float* shift  = (float*)alloc(256 * 4);

  float* h_out  = (float*)d_out;                     // [M x 256]
  float* Sh_out = h_out + (size_t)M * OUT;           // [M x 128]

  auto cdiv = [](int a, int b) { return (a + b - 1) / b; };

  // zero accumulators
  zero_kernel<<<1024, 256, 0, stream>>>(summsg, (size_t)M * OUT);
  zero_kernel<<<64, 256, 0, stream>>>(degv, (size_t)M);
  zero_kernel<<<64, 256, 0, stream>>>(sumsig, (size_t)M);
  zero_kernel<<<1, 256, 0, stream>>>(colsum, 256);
  zero_kernel<<<1, 256, 0, stream>>>(colsq, 256);

  // prep: bf16 feature + transposed bf16 weights
  cvt_bf16_kernel<<<1024, 256, 0, stream>>>(feature, featb, (size_t)M * IN);
  transpose_cvt_kernel<<<cdiv(IN * OUT, 256), 256, 0, stream>>>(W_s1, wt1, IN, OUT);
  transpose_cvt_kernel<<<cdiv(OUT * AD, 256), 256, 0, stream>>>(W_s2, wt2, OUT, AD);
  transpose_cvt_kernel<<<cdiv(AD * AD, 256), 256, 0, stream>>>(W_metric, wtm, AD, AD);
  transpose_cvt_kernel<<<cdiv(IN * OUT, 256), 256, 0, stream>>>(W_apply, wta, IN, OUT);

  // H1 = relu(feature @ W_s1 + b_s1)  [M x 256] bf16
  wmma_gemm_kernel<256, 256, true, true>
      <<<cdiv(M, 16), 256, 0, stream>>>(featb, wt1, b_s1, M, (void*)h1b);

  // Sh = softmax(H1 @ W_s2 + b_s2)  [M x 128] f32 -> d_out
  s2_softmax_kernel<<<cdiv(M, 16), 128, 0, stream>>>(h1b, wt2, b_s2, M, Sh_out);

  // per-edge Gaussian gate
  edge_metric_kernel<<<cdiv(E, 128), 256, 0, stream>>>(Sh_out, wtm, b_metric, src, dst, sigma, E);

  // gather-mul-scatter
  scatter_kernel<<<cdiv(E, 32), 256, 0, stream>>>(feature, src, dst, sigma,
                                                  summsg, degv, sumsig, E);

  // h0 (bf16, reuse h1b)
  h0_kernel<<<2048, 256, 0, stream>>>(summsg, degv, sumsig, h1b, M);

  // h = h0 @ W_apply + b_apply  [M x 256] f32 -> d_out
  wmma_gemm_kernel<256, 256, false, false>
      <<<cdiv(M, 16), 256, 0, stream>>>(h1b, wta, b_apply, M, (void*)h_out);

  // batch norm + relu (in place on d_out)
  bn_reduce_kernel<<<256, 256, 0, stream>>>(h_out, colsum, colsq, M);
  bn_finalize_kernel<<<1, 256, 0, stream>>>(colsum, colsq, bn_gamma, bn_beta, M, scale, shift);
  bn_apply_kernel<<<2048, 256, 0, stream>>>(h_out, scale, shift, (size_t)M * OUT);
}
